// RecurrentPPOAgent_42477226557880
// MI455X (gfx1250) — matrix operand order repaired and checked by reading, not compiled, not run on hardware
//
#include <hip/hip_runtime.h>
#include <hip/hip_bf16.h>

typedef __attribute__((ext_vector_type(16))) _Float16 v16h;
typedef __attribute__((ext_vector_type(8)))  _Float16 v8h;
typedef __attribute__((ext_vector_type(8)))  float    v8f;
typedef __attribute__((ext_vector_type(4)))  unsigned int u32x4;
typedef __attribute__((ext_vector_type(8)))  unsigned int u32x8;

#define BATCH      131072
#define WG_SAMPLES 64
#define WG_THREADS 128
#define USE_TDM    1

// B-fragment table in d_ws (each fragment = 32 lanes * 16 halfs = 1KB)
#define C2_FRAG   0    // conv2:  2 ntiles * 2 ksteps = 4
#define C3_FRAG   4    // conv3:  4 * 4  = 16
#define FC_FRAG   20   // fc:     4 * 2  = 8
#define WCAT_FRAG 28   // [Wih|Whh]: 24 ntiles * 7 ksteps = 168
#define A1_FRAG   196  // actor1: 4 * 4 = 16
#define C1_FRAG   212  // critic1:4 * 4 = 16
#define N_FRAGS   228

// ---------------- weight prep: f32 -> f16 B-fragments ----------------
// B fragment layout (16-bit B, K=32 x N=16, CDNA5 ISA 7.12.2):
//   lane = column N (lane&15), element i -> K = (i<8?i:i+8) + 8*(lane>>4)
__global__ void prep_weights(const float* __restrict__ c2w, const float* __restrict__ c3w,
                             const float* __restrict__ fcw, const float* __restrict__ wih,
                             const float* __restrict__ whh, const float* __restrict__ a1w,
                             const float* __restrict__ c1w, _Float16* __restrict__ ws) {
  int tid  = blockIdx.x * blockDim.x + threadIdx.x;
  int frag = tid >> 5;
  int lane = tid & 31;
  if (frag >= N_FRAGS) return;
  int nl  = lane & 15;
  int klb = (lane >> 4) * 8;
  v16h o;
  #pragma unroll
  for (int i = 0; i < 16; ++i) {
    int kl = klb + ((i < 8) ? i : i + 8);
    float v = 0.f;
    if (frag < C3_FRAG) {                       // conv2 as GEMM (32 x 64)
      int f = frag - C2_FRAG; int nt = f >> 1, ks = f & 1;
      v = c2w[(nt * 16 + nl) * 64 + ks * 32 + kl];
    } else if (frag < FC_FRAG) {                // conv3 as GEMM (64 x 128)
      int f = frag - C3_FRAG; int nt = f >> 2, ks = f & 3;
      v = c3w[(nt * 16 + nl) * 128 + ks * 32 + kl];
    } else if (frag < WCAT_FRAG) {              // fc (64 x 64)
      int f = frag - FC_FRAG; int nt = f >> 1, ks = f & 1;
      v = fcw[(nt * 16 + nl) * 64 + ks * 32 + kl];
    } else if (frag < A1_FRAG) {                // [Wih(80 pad 96) | Whh(128)] (384 x 224)
      int f = frag - WCAT_FRAG; int nt = f / 7, ks = f % 7;
      int n = nt * 16 + nl, k = ks * 32 + kl;
      if (k < 96) v = (k < 80) ? wih[n * 80 + k] : 0.f;
      else        v = whh[n * 128 + (k - 96)];
    } else if (frag < C1_FRAG) {                // actor1 (64 x 128)
      int f = frag - A1_FRAG; int nt = f >> 2, ks = f & 3;
      v = a1w[(nt * 16 + nl) * 128 + ks * 32 + kl];
    } else {                                    // critic1 (64 x 128)
      int f = frag - C1_FRAG; int nt = f >> 2, ks = f & 3;
      v = c1w[(nt * 16 + nl) * 128 + ks * 32 + kl];
    }
    o[i] = (_Float16)v;
  }
  *(v16h*)(ws + ((size_t)frag << 9) + (lane << 4)) = o;
}

// ---------------- fragment helpers ----------------
__device__ __forceinline__ v16h load_b(const _Float16* __restrict__ ws, int frag, int lane) {
  return *(const v16h*)(ws + ((size_t)frag << 9) + (lane << 4));
}
// A fragment (16-bit A, 16 x K=32): lane row = lane&15; element i -> K = (i<8?i:i+8)+8*(lane>>4)
__device__ __forceinline__ v16h load_a(const _Float16* base, int pitch, int row, int kstart, int h) {
  const _Float16* p = base + (size_t)row * pitch + kstart + 8 * h;
  v8h lo = *(const v8h*)p;
  v8h hi = *(const v8h*)(p + 16);
  return __builtin_shufflevector(lo, hi, 0,1,2,3,4,5,6,7,8,9,10,11,12,13,14,15);
}
__device__ __forceinline__ v8f wmma16(v16h a, v16h b, v8f c) {
  return __builtin_amdgcn_wmma_f32_16x16x32_f16(false, a, false, b, (short)0, c, false, false);
}
__device__ __forceinline__ float sigf(float x) { return 1.f / (1.f + __expf(-x)); }

// ---------------- LDS region map (bytes), regions reused across phases ----------------
#define OBS_OFF   0       // [64][147] f32  = 37632   (conv1 in, TDM dest, pitch 147)
#define X_OFF     0       // [64][72]  f16  = 9216    (conv3 out / fc in)
#define INP_OFF   9216    // [64][232] f16  = 29696   (fc/embed/hidden -> GRU A)
#define H_OFF     9216    // [64][136] f16  = 17408   (h_new -> heads A)
#define POOL_OFF  37888   // [64][144] f16  = 18432   (conv1 out / conv2 in)
#define RZ_OFF    40960   // [64][264] f16  = 33792   (GRU r,z preacts)
#define HA_OFF    40960   // [64][68]  f32  = 17408   (tanh actor hidden)
#define X2_OFF    56320   // [64][136] f16  = 17408   (conv2 out / conv3 A)
#define HC_OFF    58368   // [64][68]  f32  = 17408   (tanh critic hidden)
#define XN_OFF    74752   // [64][136] f16  = 17408   (GRU xn)
#define HN_OFF    92160   // [64][136] f16  = 17408   (GRU hn)
#define SMEM_SIZE 109568

__global__ void __launch_bounds__(WG_THREADS)
agent_fused(const float* __restrict__ obs, const int* __restrict__ prev_action,
            const float* __restrict__ hidden,
            const float* __restrict__ conv1_w, const float* __restrict__ conv1_b,
            const float* __restrict__ conv2_b, const float* __restrict__ conv3_b,
            const float* __restrict__ fc_b,    const float* __restrict__ act_embed,
            const float* __restrict__ gru_bih, const float* __restrict__ gru_bhh,
            const float* __restrict__ a1_b,    const float* __restrict__ a2_w,
            const float* __restrict__ a2_b,    const float* __restrict__ c1_b,
            const float* __restrict__ c2_w,    const float* __restrict__ c2_b,
            const _Float16* __restrict__ ws,   float* __restrict__ out) {
  __shared__ __align__(32) unsigned char smem[SMEM_SIZE];
  float*    obs_s  = (float*)   (smem + OBS_OFF);
  _Float16* pool_s = (_Float16*)(smem + POOL_OFF);
  _Float16* x2_s   = (_Float16*)(smem + X2_OFF);
  _Float16* x_s    = (_Float16*)(smem + X_OFF);
  _Float16* inp_s  = (_Float16*)(smem + INP_OFF);
  _Float16* rz_s   = (_Float16*)(smem + RZ_OFF);
  _Float16* xn_s   = (_Float16*)(smem + XN_OFF);
  _Float16* hn_s   = (_Float16*)(smem + HN_OFF);
  _Float16* h_s    = (_Float16*)(smem + H_OFF);
  float*    ha_s   = (float*)   (smem + HA_OFF);
  float*    hc_s   = (float*)   (smem + HC_OFF);

  const int t     = threadIdx.x;
  const int lane  = t & 31;
  const int wid   = t >> 5;
  const int nl    = lane & 15;
  const int hh    = lane >> 4;
  const int sBase = wid * 16;                 // wave's 16-sample tile (conv/fc phases)
  const int gBase = blockIdx.x * WG_SAMPLES;  // global sample base

  // ---- phase 0: stage obs tile (37632B global -> LDS) via Tensor Data Mover ----
#if USE_TDM
  if (wid == 0) {
    size_t ga = (size_t)(const void*)(obs + (size_t)gBase * 147);
    unsigned ldsoff = (unsigned)(size_t)(void*)obs_s;  // ISA 10.2: LDS flat addr[31:0] == LDS offset
    const unsigned ndw = WG_SAMPLES * 147;             // 9408 dwords, 1-D tile
    u32x4 g0;
    g0[0] = 1u;                                        // count=1, load, no gather
    g0[1] = ldsoff;                                    // lds_addr
    g0[2] = (unsigned)(ga & 0xffffffffu);              // global_addr[31:0]
    g0[3] = (unsigned)((ga >> 32) & 0x01ffffffu) | (2u << 30);  // addr[56:32] | type=2
    u32x8 g1;
    g1[0] = 2u << 16;                                  // data_size=4B; mask=0; no pad/iter
    g1[1] = (ndw & 0xffffu) << 16;                     // tensor_dim0[15:0]
    g1[2] = (1u << 16) | (ndw >> 16);                  // tensor_dim0[31:16] | tensor_dim1=1
    g1[3] = (ndw & 0xffffu) << 16;                     // tensor_dim1[31:16]=0 | tile_dim0
    g1[4] = 1u;                                        // tile_dim1=1, tile_dim2=0
    g1[5] = ndw;                                       // tensor_dim0_stride[31:0]
    g1[6] = 0u;
    g1[7] = 0u;
    asm volatile("tensor_load_to_lds %0, %1" :: "s"(g0), "s"(g1) : "memory");
    __builtin_amdgcn_s_wait_tensorcnt(0);
  }
#else
  for (int idx = t; idx < WG_SAMPLES * 147; idx += WG_THREADS) {
    int s = idx / 147, j = idx - s * 147;
    obs_s[s * 147 + j] = obs[(size_t)(gBase + s) * 147 + j];
  }
#endif
  // warm L2/WGP$ with the GRU weight fragments (dominant phase-5 stream)
  for (unsigned off = (unsigned)t * 128u; off < (unsigned)N_FRAGS * 1024u;
       off += WG_THREADS * 128u)
    __builtin_prefetch((const char*)ws + off, 0, 1);
  __syncthreads();

  // ---- phase 1: conv1 (3->16, 2x2) + relu + 2x2 maxpool, scalar; 2 threads/sample ----
  {
    int s = t & 63, half = t >> 6;
    const float* row = obs_s + s * 147;  // img[c][y][x] = row[(y*7+x)*3+c]
    for (int o8 = 0; o8 < 8; ++o8) {
      int oc = half * 8 + o8;
      float w[12];
      #pragma unroll
      for (int q = 0; q < 12; ++q) w[q] = conv1_w[oc * 12 + q];
      float bias = conv1_b[oc];
      for (int py = 0; py < 3; ++py)
        for (int px = 0; px < 3; ++px) {
          float mx = -3.0e38f;
          #pragma unroll
          for (int dy = 0; dy < 2; ++dy)
            #pragma unroll
            for (int dx = 0; dx < 2; ++dx) {
              int y = 2 * py + dy, x = 2 * px + dx;
              float acc = bias;
              #pragma unroll
              for (int c = 0; c < 3; ++c)
                #pragma unroll
                for (int ky = 0; ky < 2; ++ky)
                  #pragma unroll
                  for (int kx = 0; kx < 2; ++kx)
                    acc = fmaf(w[c * 4 + ky * 2 + kx],
                               row[((y + ky) * 7 + (x + kx)) * 3 + c], acc);
              mx = fmaxf(mx, acc);
            }
          pool_s[s * 144 + oc * 9 + py * 3 + px] = (_Float16)fmaxf(mx, 0.f);
        }
    }
  }
  __syncthreads();

  // ---- phase 2: conv2 (16->32, 2x2) as 4 position-GEMMs, K=64, WMMA ----
  for (int p = 0; p < 4; ++p) {
    int py = p >> 1, px = p & 1;
    v16h a[2];
    for (int ks = 0; ks < 2; ++ks) {
      v16h av;
      #pragma unroll
      for (int i = 0; i < 16; ++i) {   // im2col gather per ISA A-pattern
        int k = ks * 32 + ((i < 8) ? i : i + 8) + 8 * hh;
        int c = k >> 2, r4 = k & 3, ky = r4 >> 1, kx = r4 & 1;
        av[i] = pool_s[(sBase + nl) * 144 + c * 9 + (py + ky) * 3 + (px + kx)];
      }
      a[ks] = av;
    }
    for (int nt = 0; nt < 2; ++nt) {
      v8f acc = {};
      acc = wmma16(a[0], load_b(ws, C2_FRAG + nt * 2 + 0, lane), acc);
      acc = wmma16(a[1], load_b(ws, C2_FRAG + nt * 2 + 1, lane), acc);
      int o = nt * 16 + nl;
      float b = conv2_b[o];
      #pragma unroll
      for (int j = 0; j < 8; ++j)
        x2_s[(sBase + j + 8 * hh) * 136 + o * 4 + p] = (_Float16)fmaxf(acc[j] + b, 0.f);
    }
  }
  __syncthreads();

  // ---- phase 3: conv3 (32->64, 2x2) == GEMM (B,128)x(128,64), WMMA ----
  {
    v16h a[4];
    for (int ks = 0; ks < 4; ++ks) a[ks] = load_a(x2_s, 136, sBase + nl, ks * 32, hh);
    for (int nt = 0; nt < 4; ++nt) {
      v8f acc = {};
      for (int ks = 0; ks < 4; ++ks)
        acc = wmma16(a[ks], load_b(ws, C3_FRAG + nt * 4 + ks, lane), acc);
      int o = nt * 16 + nl;
      float b = conv3_b[o];
      #pragma unroll
      for (int j = 0; j < 8; ++j)
        x_s[(sBase + j + 8 * hh) * 72 + o] = (_Float16)fmaxf(acc[j] + b, 0.f);
    }
  }
  __syncthreads();

  // ---- phase 4: build GRU input [obs_enc(64) | act_emb(16) | pad(16) | hidden(128)] ----
  if (t < 64) {
    int s = t;
    int aidx = prev_action[gBase + s];
    #pragma unroll
    for (int j = 0; j < 16; ++j) inp_s[s * 232 + 64 + j] = (_Float16)act_embed[aidx * 16 + j];
    #pragma unroll
    for (int j = 0; j < 16; ++j) inp_s[s * 232 + 80 + j] = (_Float16)0.f;
  }
  for (int idx = t; idx < WG_SAMPLES * 128; idx += WG_THREADS) {
    int s = idx >> 7, f = idx & 127;
    inp_s[s * 232 + 96 + f] = (_Float16)hidden[(size_t)(gBase + s) * 128 + f];
  }
  {  // fc: obs_enc = relu(x @ fcW^T + b), K=64, WMMA
    v16h a[2];
    for (int ks = 0; ks < 2; ++ks) a[ks] = load_a(x_s, 72, sBase + nl, ks * 32, hh);
    for (int nt = 0; nt < 4; ++nt) {
      v8f acc = {};
      for (int ks = 0; ks < 2; ++ks)
        acc = wmma16(a[ks], load_b(ws, FC_FRAG + nt * 2 + ks, lane), acc);
      int n = nt * 16 + nl;
      float b = fc_b[n];
      #pragma unroll
      for (int j = 0; j < 8; ++j)
        inp_s[(sBase + j + 8 * hh) * 232 + n] = (_Float16)fmaxf(acc[j] + b, 0.f);
    }
  }
  __syncthreads();

  // ---- phase 5: GRU gate pre-activations, concatenated K=224 (7 ksteps), WMMA ----
  // Each wave owns N-tiles {wid, wid+4, ...}: B fragments loaded ONCE into registers,
  // then swept across all four 16-row A tiles (4x cut in global fragment traffic).
  for (int nt = wid; nt < 24; nt += 4) {
    v16h b[7];
    #pragma unroll
    for (int ks = 0; ks < 7; ++ks) b[ks] = load_b(ws, WCAT_FRAG + nt * 7 + ks, lane);
    if (nt < 16) {                      // r and z gates: gx+gh fused accumulation
      int n = nt * 16 + nl;
      float bsum = gru_bih[n] + gru_bhh[n];
      for (int rt = 0; rt < 4; ++rt) {
        int rb = rt * 16;
        v8f acc = {};
        for (int ks = 0; ks < 7; ++ks)
          acc = wmma16(load_a(inp_s, 232, rb + nl, ks * 32, hh), b[ks], acc);
        #pragma unroll
        for (int j = 0; j < 8; ++j)
          rz_s[(rb + j + 8 * hh) * 264 + n] = (_Float16)(acc[j] + bsum);
      }
    } else {                            // n gate: xn, hn kept separate
      int n = nt * 16 + nl, f = n - 256;
      float bx = gru_bih[n], bh = gru_bhh[n];
      for (int rt = 0; rt < 4; ++rt) {
        int rb = rt * 16;
        v8f accx = {}, acch = {};
        for (int ks = 0; ks < 3; ++ks)
          accx = wmma16(load_a(inp_s, 232, rb + nl, ks * 32, hh), b[ks], accx);
        for (int ks = 3; ks < 7; ++ks)
          acch = wmma16(load_a(inp_s, 232, rb + nl, ks * 32, hh), b[ks], acch);
        #pragma unroll
        for (int j = 0; j < 8; ++j) {
          int srow = rb + j + 8 * hh;
          xn_s[srow * 136 + f] = (_Float16)(accx[j] + bx);
          hn_s[srow * 136 + f] = (_Float16)(acch[j] + bh);
        }
      }
    }
  }
  __syncthreads();

  // ---- phase 6: gate nonlinearity + h_new (VALU), write h out + stage f16 ----
  {
    int s = t & 63, half = t >> 6;
    size_t gs = (size_t)(gBase + s);
    for (int f0 = 0; f0 < 64; ++f0) {
      int f = half * 64 + f0;
      float r = sigf((float)rz_s[s * 264 + f]);
      float z = sigf((float)rz_s[s * 264 + 128 + f]);
      float n = tanhf((float)xn_s[s * 136 + f] + r * (float)hn_s[s * 136 + f]);
      float h = (1.f - z) * n + z * hidden[gs * 128 + f];
      out[(size_t)BATCH * 8 + gs * 128 + f] = h;
      h_s[s * 136 + f] = (_Float16)h;
    }
  }
  __syncthreads();

  // ---- phase 7: actor/critic hidden layers, K=128, WMMA + tanh ----
  {
    v16h a[4];
    for (int ks = 0; ks < 4; ++ks) a[ks] = load_a(h_s, 136, sBase + nl, ks * 32, hh);
    for (int nt = 0; nt < 4; ++nt) {
      v8f acc = {};
      for (int ks = 0; ks < 4; ++ks)
        acc = wmma16(a[ks], load_b(ws, A1_FRAG + nt * 4 + ks, lane), acc);
      int n = nt * 16 + nl;
      float b = a1_b[n];
      #pragma unroll
      for (int j = 0; j < 8; ++j)
        ha_s[(sBase + j + 8 * hh) * 68 + n] = tanhf(acc[j] + b);
    }
    for (int nt = 0; nt < 4; ++nt) {
      v8f acc = {};
      for (int ks = 0; ks < 4; ++ks)
        acc = wmma16(a[ks], load_b(ws, C1_FRAG + nt * 4 + ks, lane), acc);
      int n = nt * 16 + nl;
      float b = c1_b[n];
      #pragma unroll
      for (int j = 0; j < 8; ++j)
        hc_s[(sBase + j + 8 * hh) * 68 + n] = tanhf(acc[j] + b);
    }
  }
  __syncthreads();

  // ---- phase 8: tiny output heads (7x64 and 1x64 matvecs), scalar ----
  {
    int s = t & 63;
    size_t gs = (size_t)(gBase + s);
    if (t < 64) {
      for (int j = 0; j < 7; ++j) {
        float acc = a2_b[j];
        for (int k = 0; k < 64; ++k) acc = fmaf(ha_s[s * 68 + k], a2_w[j * 64 + k], acc);
        out[gs * 7 + j] = acc;
      }
    } else {
      float acc = c2_b[0];
      for (int k = 0; k < 64; ++k) acc = fmaf(hc_s[s * 68 + k], c2_w[k], acc);
      out[(size_t)BATCH * 7 + gs] = acc;
    }
  }
}

extern "C" void kernel_launch(void* const* d_in, const int* in_sizes, int n_in,
                              void* d_out, int out_size, void* d_ws, size_t ws_size,
                              hipStream_t stream) {
  (void)in_sizes; (void)n_in; (void)out_size; (void)ws_size;
  const float* obs       = (const float*)d_in[0];
  const int*   pact      = (const int*)  d_in[1];
  const float* hidden    = (const float*)d_in[2];
  const float* conv1_w   = (const float*)d_in[3];
  const float* conv1_b   = (const float*)d_in[4];
  const float* conv2_w   = (const float*)d_in[5];
  const float* conv2_b   = (const float*)d_in[6];
  const float* conv3_w   = (const float*)d_in[7];
  const float* conv3_b   = (const float*)d_in[8];
  const float* fc_w      = (const float*)d_in[9];
  const float* fc_b      = (const float*)d_in[10];
  const float* act_embed = (const float*)d_in[11];
  const float* gru_wih   = (const float*)d_in[12];
  const float* gru_whh   = (const float*)d_in[13];
  const float* gru_bih   = (const float*)d_in[14];
  const float* gru_bhh   = (const float*)d_in[15];
  const float* a1_w      = (const float*)d_in[16];
  const float* a1_b      = (const float*)d_in[17];
  const float* a2_w      = (const float*)d_in[18];
  const float* a2_b      = (const float*)d_in[19];
  const float* c1_w      = (const float*)d_in[20];
  const float* c1_b      = (const float*)d_in[21];
  const float* c2_w      = (const float*)d_in[22];
  const float* c2_b      = (const float*)d_in[23];
  _Float16* ws  = (_Float16*)d_ws;
  float*    out = (float*)d_out;

  prep_weights<<<(N_FRAGS * 32 + 127) / 128, 128, 0, stream>>>(
      conv2_w, conv3_w, fc_w, gru_wih, gru_whh, a1_w, c1_w, ws);

  agent_fused<<<BATCH / WG_SAMPLES, WG_THREADS, 0, stream>>>(
      obs, pact, hidden, conv1_w, conv1_b, conv2_b, conv3_b, fc_b, act_embed,
      gru_bih, gru_bhh, a1_b, a2_w, a2_b, c1_b, c2_w, c2_b, ws, out);
}
